// GAT_17420387352864
// MI455X (gfx1250) — matrix operand order repaired and checked by reading, not compiled, not run on hardware
//
#include <hip/hip_runtime.h>
#include <hip/hip_bf16.h>

typedef __attribute__((ext_vector_type(16))) __bf16 v16bf;
typedef __attribute__((ext_vector_type(8)))  float  v8f;

constexpr int NB = 4096;  // batch
constexpr int DD = 64;    // input channels
constexpr int TT = 60;    // timesteps
constexpr int HH = 128;   // hidden
constexpr int GG = 512;   // 4*H

// ---------------- workspace layout (bytes) ----------------
constexpr size_t OFF_SCALE1 = 0;                               // 64 f32
constexpr size_t OFF_SHIFT1 = OFF_SCALE1 + 64 * 4;             // 64 f32
constexpr size_t OFF_BSUM   = OFF_SHIFT1 + 64 * 4;             // 512 f32
constexpr size_t OFF_U1     = OFF_BSUM + GG * 4;               // 128 f32
constexpr size_t OFF_U2     = OFF_U1 + HH * 4;                 // 128 f32
constexpr size_t OFF_C12    = OFF_U2 + HH * 4;                 // 2 f32
constexpr size_t OFF_SCALE2 = OFF_C12 + 16;                    // 128 f32
constexpr size_t OFF_SHIFT2 = OFF_SCALE2 + HH * 4;             // 128 f32
constexpr size_t OFF_WIH    = (OFF_SHIFT2 + HH * 4 + 255) & ~size_t(255); // [512][64] bf16
constexpr size_t OFF_WHH    = OFF_WIH + (size_t)GG * DD * 2;   // [512][128] bf16
constexpr size_t OFF_WFC    = OFF_WHH + (size_t)GG * HH * 2;   // [128][128] bf16
constexpr size_t OFF_HID    = OFF_WFC + (size_t)HH * HH * 2;   // [N][H] f32
constexpr size_t OFF_HIDB   = OFF_HID + (size_t)NB * HH * 4;   // [N][H] bf16
constexpr size_t OFF_S1     = OFF_HIDB + (size_t)NB * HH * 2;  // N f32
constexpr size_t OFF_S2     = OFF_S1 + (size_t)NB * 4;         // N f32
constexpr size_t OFF_MAX    = OFF_S2 + (size_t)NB * 4;         // N f32
constexpr size_t OFF_INVL   = OFF_MAX + (size_t)NB * 4;        // N f32

// ---------------- helpers ----------------
union Frag { v16bf v; unsigned int u[8]; };

__device__ __forceinline__ unsigned short f2bf_u16(float f) {
  union { float f; unsigned int u; } x; x.f = f;
  return (unsigned short)((x.u + 0x7fffu + ((x.u >> 16) & 1u)) >> 16);
}

__device__ __forceinline__ int a_kbase(int v, int half) {
  // 16-bit A-matrix 16x32, contiguous k order (matches B fills): VGPR v holds K pair 16*half + 2*v
  return 16 * half + 2 * v;
}

__device__ __forceinline__ float sigm(float z) { return 1.f / (1.f + __expf(-z)); }
__device__ __forceinline__ float leaky(float z) { return z < 0.f ? 0.01f * z : z; }

// low 32 bits of a generic LDS pointer == LDS byte address (ISA 10.2 aperture rules)
__device__ __forceinline__ unsigned lds_off(const void* p) {
  return (unsigned)(unsigned long long)p;
}

// CDNA5 async global->LDS DMA (GVS mode), tracked by ASYNCcnt (ISA 15.18.3 op 98)
__device__ __forceinline__ void async_ld_b128(unsigned lds_dst, unsigned byte_off,
                                              const void* base) {
  asm volatile("global_load_async_to_lds_b128 %0, %1, %2"
               :: "v"(lds_dst), "v"(byte_off), "s"(base)
               : "memory");
}
__device__ __forceinline__ void wait_async0() {
  asm volatile("s_wait_asynccnt 0x0" ::: "memory");
}

// ---------------- K1: BN1 stats over (N,T) per channel d ----------------
__global__ __launch_bounds__(256) void bn1_stats_kernel(const float* __restrict__ x,
                                                        const float* __restrict__ g1,
                                                        const float* __restrict__ b1,
                                                        char* __restrict__ ws) {
  const int d = blockIdx.x;
  const int tid = threadIdx.x;
  float s = 0.f, ss = 0.f;
  for (int i = tid; i < NB * TT; i += 256) {
    int n = i / TT, t = i - n * TT;
    float v = x[(size_t)n * (DD * TT) + (size_t)d * TT + t];
    s += v; ss += v * v;
  }
  __shared__ float r1[256], r2[256];
  r1[tid] = s; r2[tid] = ss;
  __syncthreads();
  for (int o = 128; o > 0; o >>= 1) {
    if (tid < o) { r1[tid] += r1[tid + o]; r2[tid] += r2[tid + o]; }
    __syncthreads();
  }
  if (tid == 0) {
    float mean = r1[0] / (float)(NB * TT);
    float var  = r2[0] / (float)(NB * TT) - mean * mean;
    float sc = g1[d] * rsqrtf(var + 1e-5f);
    ((float*)(ws + OFF_SCALE1))[d] = sc;
    ((float*)(ws + OFF_SHIFT1))[d] = b1[d] - mean * sc;
  }
}

// ---------------- K2: weight conversions + algebraic folds ----------------
__global__ __launch_bounds__(256) void prep_kernel(const float* __restrict__ Wih,
                                                   const float* __restrict__ Whh,
                                                   const float* __restrict__ bih,
                                                   const float* __restrict__ bhh,
                                                   const float* __restrict__ Wt,
                                                   const float* __restrict__ bt,
                                                   const float* __restrict__ av,
                                                   const float* __restrict__ Wfc,
                                                   char* __restrict__ ws) {
  const int tid = blockIdx.x * blockDim.x + threadIdx.x;
  const int stride = gridDim.x * blockDim.x;
  unsigned short* wih = (unsigned short*)(ws + OFF_WIH);
  unsigned short* whh = (unsigned short*)(ws + OFF_WHH);
  unsigned short* wfc = (unsigned short*)(ws + OFF_WFC);
  float* bs = (float*)(ws + OFF_BSUM);
  float* u1 = (float*)(ws + OFF_U1);
  float* u2 = (float*)(ws + OFF_U2);
  float* c12 = (float*)(ws + OFF_C12);

  for (int i = tid; i < GG * DD; i += stride) wih[i] = f2bf_u16(Wih[i]);
  for (int i = tid; i < GG * HH; i += stride) whh[i] = f2bf_u16(Whh[i]);
  for (int i = tid; i < HH * HH; i += stride) wfc[i] = f2bf_u16(Wfc[i]);
  for (int i = tid; i < GG; i += stride) bs[i] = bih[i] + bhh[i];
  // s1 = hidden @ u1 + c1 ; u1[j] = sum_k W_t[k,j]*a[k]  (fold W_t into a)
  for (int j = tid; j < HH; j += stride) {
    float a1 = 0.f, a2 = 0.f;
    for (int k = 0; k < HH; ++k) {
      float w = Wt[k * HH + j];
      a1 += w * av[k];
      a2 += w * av[HH + k];
    }
    u1[j] = a1; u2[j] = a2;
  }
  if (tid == 0) {
    float c1 = 0.f, c2 = 0.f;
    for (int k = 0; k < HH; ++k) { c1 += bt[k] * av[k]; c2 += bt[k] * av[HH + k]; }
    c12[0] = c1; c12[1] = c2;
  }
}

// ---------------- K3: fused BN1-apply + LSTM (recurrence stays on-chip) ----------------
// 256 blocks x 256 threads (8 waves). Each block owns 16 rows of N.
// LDS: Wih [512][64] bf16, Whh [512][128] bf16 (async-staged once), As [16][64] bf16,
//      Hs [16][128] bf16, Gs [16][512] f32  => 235,520 bytes of the 320KB WGP LDS.
constexpr unsigned LSTM_LDS =
    (unsigned)(GG * DD * 2 + GG * HH * 2 + 16 * DD * 2 + 16 * HH * 2 + 16 * GG * 4);

__global__ __launch_bounds__(256) void lstm_kernel(const float* __restrict__ x,
                                                   char* __restrict__ ws) {
  extern __shared__ char smem[];
  unsigned short* Wih_s = (unsigned short*)smem;                        // [512][64]
  unsigned short* Whh_s = (unsigned short*)(smem + GG * DD * 2);        // [512][128]
  unsigned short* As    = (unsigned short*)(smem + GG * DD * 2 + GG * HH * 2);      // [16][64]
  unsigned short* Hs    = (unsigned short*)((char*)As + 16 * DD * 2);   // [16][128]
  float*          Gs    = (float*)((char*)Hs + 16 * HH * 2);            // [16][512]

  const float* scale1 = (const float*)(ws + OFF_SCALE1);
  const float* shift1 = (const float*)(ws + OFF_SHIFT1);
  const float* bsum   = (const float*)(ws + OFF_BSUM);
  float* hid = (float*)(ws + OFF_HID);

  const int tid  = threadIdx.x;
  const int wave = tid >> 5;
  const int lane = tid & 31;
  const int half = lane >> 4;
  const int l16  = lane & 15;
  const int n0   = blockIdx.x * 16;
  const int gbase = wave * 64;   // 64 gate columns per wave (4 n-tiles)

  // async-stage both weight blocks (contiguous 192KB in ws and in LDS) once
  {
    const void* wsrc = (const void*)(ws + OFF_WIH);
    const unsigned wdst = lds_off(Wih_s);
    for (int e = tid; e < (GG * DD * 2 + GG * HH * 2) / 16; e += 256)
      async_ld_b128(wdst + (unsigned)e * 16u, (unsigned)e * 16u, wsrc);
  }
  for (int i = tid; i < 16 * HH; i += 256) Hs[i] = (unsigned short)0;
  wait_async0();

  // cell state in registers: element e = tid + k*256 -> (row=e/128, col=e%128)
  float cst[8];
#pragma unroll
  for (int k = 0; k < 8; ++k) cst[k] = 0.f;

  for (int t = 0; t < TT; ++t) {
    __syncthreads();
    // BN1-normalized x slice for this timestep -> bf16 A tile
    for (int e = tid; e < 16 * DD; e += 256) {
      int r = e >> 6, d = e & 63;
      float v = x[(size_t)(n0 + r) * (DD * TT) + (size_t)d * TT + t];
      As[r * DD + d] = f2bf_u16(v * scale1[d] + shift1[d]);
    }
    __syncthreads();

    v8f acc[4];
#pragma unroll
    for (int nt = 0; nt < 4; ++nt) {
      float b = bsum[gbase + nt * 16 + l16];
#pragma unroll
      for (int r = 0; r < 8; ++r) acc[nt][r] = b;
    }

    // K = 64 (x part, 2 kfrags) + 128 (h part, 4 kfrags)
#pragma unroll
    for (int kf = 0; kf < 6; ++kf) {
      const unsigned short* Asrc = (kf < 2) ? As : Hs;
      const unsigned short* Bsrc = (kf < 2) ? Wih_s : Whh_s;
      const int ldk = (kf < 2) ? DD : HH;
      const int k0  = (kf < 2) ? kf * 32 : (kf - 2) * 32;
      Frag af;
#pragma unroll
      for (int v = 0; v < 8; ++v)
        af.u[v] = *(const unsigned int*)&Asrc[l16 * ldk + k0 + a_kbase(v, half)];
#pragma unroll
      for (int nt = 0; nt < 4; ++nt) {
        Frag bf;
        const unsigned short* bp = &Bsrc[(size_t)(gbase + nt * 16 + l16) * ldk + k0 + half * 16];
#pragma unroll
        for (int v = 0; v < 8; ++v) bf.u[v] = *(const unsigned int*)&bp[2 * v];
        acc[nt] = __builtin_amdgcn_wmma_f32_16x16x32_bf16(
            false, af.v, false, bf.v, (short)0, acc[nt], false, false);
      }
    }

    // spill gates to LDS for the cross-wave i/f/g/o combine
#pragma unroll
    for (int nt = 0; nt < 4; ++nt)
#pragma unroll
      for (int r = 0; r < 8; ++r)
        Gs[(size_t)(r + 8 * half) * GG + gbase + nt * 16 + l16] = acc[nt][r];
    __syncthreads();

    // c,h update: 2048 elements / 256 threads
#pragma unroll
    for (int k = 0; k < 8; ++k) {
      int e = tid + k * 256;
      int r = e >> 7, c = e & 127;
      float gi = sigm(Gs[(size_t)r * GG + c]);
      float gf = sigm(Gs[(size_t)r * GG + 128 + c]);
      float gg = tanhf(Gs[(size_t)r * GG + 256 + c]);
      float go = sigm(Gs[(size_t)r * GG + 384 + c]);
      float cn = gf * cst[k] + gi * gg;
      cst[k] = cn;
      float h = go * tanhf(cn);
      Hs[r * HH + c] = f2bf_u16(h);
      if (t == TT - 1) hid[(size_t)(n0 + r) * HH + c] = h;
    }
  }
}

// ---------------- K4: BN2 stats per hidden column ----------------
__global__ __launch_bounds__(256) void bn2_stats_kernel(const float* __restrict__ g2,
                                                        const float* __restrict__ b2,
                                                        char* __restrict__ ws) {
  const int c = blockIdx.x;
  const int tid = threadIdx.x;
  const float* hid = (const float*)(ws + OFF_HID);
  float s = 0.f, ss = 0.f;
  for (int n = tid; n < NB; n += 256) {
    float v = hid[(size_t)n * HH + c];
    s += v; ss += v * v;
  }
  __shared__ float r1[256], r2[256];
  r1[tid] = s; r2[tid] = ss;
  __syncthreads();
  for (int o = 128; o > 0; o >>= 1) {
    if (tid < o) { r1[tid] += r1[tid + o]; r2[tid] += r2[tid + o]; }
    __syncthreads();
  }
  if (tid == 0) {
    float mean = r1[0] / (float)NB;
    float var  = r2[0] / (float)NB - mean * mean;
    float sc = g2[c] * rsqrtf(var + 1e-5f);
    ((float*)(ws + OFF_SCALE2))[c] = sc;
    ((float*)(ws + OFF_SHIFT2))[c] = b2[c] - mean * sc;
  }
}

// ---------------- K5: apply BN2, emit bf16 hidden, per-row s1/s2 ----------------
__global__ __launch_bounds__(128) void rownorm_kernel(char* __restrict__ ws) {
  const int n = blockIdx.x, c = threadIdx.x;
  float* hid = (float*)(ws + OFF_HID);
  unsigned short* hidb = (unsigned short*)(ws + OFF_HIDB);
  const float* scale2 = (const float*)(ws + OFF_SCALE2);
  const float* shift2 = (const float*)(ws + OFF_SHIFT2);
  const float* u1 = (const float*)(ws + OFF_U1);
  const float* u2 = (const float*)(ws + OFF_U2);
  const float* c12 = (const float*)(ws + OFF_C12);

  float v = hid[(size_t)n * HH + c] * scale2[c] + shift2[c];
  hid[(size_t)n * HH + c] = v;
  hidb[(size_t)n * HH + c] = f2bf_u16(v);

  __shared__ float r1[128], r2[128];
  r1[c] = v * u1[c]; r2[c] = v * u2[c];
  __syncthreads();
  for (int o = 64; o > 0; o >>= 1) {
    if (c < o) { r1[c] += r1[c + o]; r2[c] += r2[c + o]; }
    __syncthreads();
  }
  if (c == 0) {
    ((float*)(ws + OFF_S1))[n] = r1[0] + c12[0];
    ((float*)(ws + OFF_S2))[n] = r2[0] + c12[1];
  }
}

// ---------------- K6: softmax row statistics (two-pass) ----------------
__global__ __launch_bounds__(256) void softmax_stats_kernel(char* __restrict__ ws) {
  __shared__ float s1s[NB];   // 16 KB
  const float* s1 = (const float*)(ws + OFF_S1);
  const float* s2 = (const float*)(ws + OFF_S2);
  float* mM   = (float*)(ws + OFF_MAX);
  float* invl = (float*)(ws + OFF_INVL);
  const int tid = threadIdx.x;
  for (int i = tid; i < NB; i += 256) s1s[i] = s1[i];
  __syncthreads();
  const int i = blockIdx.x * 256 + tid;
  const float s2i = s2[i];
  float m = -3.4e38f;
  for (int j = 0; j < NB; ++j) m = fmaxf(m, leaky(s2i + s1s[j]));
  float l = 0.f;
  for (int j = 0; j < NB; ++j) l += __expf(leaky(s2i + s1s[j]) - m);
  mM[i] = m;
  invl[i] = 1.f / l;
}

// ---------------- K7: attention GEMM + residual + FC + sigmoid ----------------
// 32 blocks x 8 waves; wave owns a 16-row tile. hidden tiles are DOUBLE-BUFFERED
// and staged with global_load_async_to_lds_b128 (ASYNCcnt), overlapping the DMA of
// tile jt+64 with the 16 WMMAs on tile jt.
// LDS: Hb0/Hb1 16KB each + s1c0/s1c1 + Wfc 32KB + h2 64KB = 131,584 B.
constexpr unsigned HB_BYTES      = 64 * HH * 2;                     // 16384
constexpr unsigned ATTN_OFF_HB0  = 0;
constexpr unsigned ATTN_OFF_HB1  = HB_BYTES;
constexpr unsigned ATTN_OFF_S1C0 = 2 * HB_BYTES;
constexpr unsigned ATTN_OFF_S1C1 = ATTN_OFF_S1C0 + 256;
constexpr unsigned ATTN_OFF_WFC  = ATTN_OFF_S1C1 + 256;
constexpr unsigned ATTN_OFF_H2   = ATTN_OFF_WFC + HH * HH * 2;
constexpr unsigned ATTN_LDS      = ATTN_OFF_H2 + 8 * 16 * HH * 4;   // 131,584

__global__ __launch_bounds__(256) void attn_out_kernel(const float* __restrict__ bfc,
                                                       const float* __restrict__ Wout,
                                                       const float* __restrict__ bout,
                                                       char* __restrict__ ws,
                                                       float* __restrict__ out) {
  extern __shared__ char smem[];
  unsigned short* const Hb0 = (unsigned short*)(smem + ATTN_OFF_HB0);
  unsigned short* const Hb1 = (unsigned short*)(smem + ATTN_OFF_HB1);
  float* const s1c0 = (float*)(smem + ATTN_OFF_S1C0);
  float* const s1c1 = (float*)(smem + ATTN_OFF_S1C1);
  unsigned short* Wfc_s = (unsigned short*)(smem + ATTN_OFF_WFC);
  float* h2all = (float*)(smem + ATTN_OFF_H2);

  const int tid  = threadIdx.x;
  const int wave = tid >> 5;
  const int lane = tid & 31;
  const int half = lane >> 4;
  const int l16  = lane & 15;
  float* h2s = h2all + (size_t)wave * 16 * HH;

  const float* hid  = (const float*)(ws + OFF_HID);
  const unsigned short* hidb = (const unsigned short*)(ws + OFF_HIDB);
  const float* s1   = (const float*)(ws + OFF_S1);
  const float* s2   = (const float*)(ws + OFF_S2);
  const float* mM   = (const float*)(ws + OFF_MAX);
  const float* invl = (const float*)(ws + OFF_INVL);
  const unsigned short* WfcG = (const unsigned short*)(ws + OFF_WFC);

  const unsigned hb0o = lds_off(Hb0);
  const unsigned hb1o = lds_off(Hb1);

  for (int i = tid; i < HH * HH / 2; i += 256)
    ((unsigned int*)Wfc_s)[i] = ((const unsigned int*)WfcG)[i];

  const int i0 = (blockIdx.x * 8 + wave) * 16;
  const int irow = i0 + l16;
  const float s2i = s2[irow];
  const float mi  = mM[irow];
  const float li  = invl[irow];

  v8f acc[8];
#pragma unroll
  for (int nt = 0; nt < 8; ++nt)
#pragma unroll
    for (int r = 0; r < 8; ++r) acc[nt][r] = 0.f;

  // prime buffer 0 (64 rows x 256B = 1024 b128 transfers, 4 per thread)
  for (int e = tid; e < (int)(HB_BYTES / 16); e += 256)
    async_ld_b128(hb0o + (unsigned)e * 16u, (unsigned)e * 16u, (const void*)hidb);
  if (tid < 64) s1c0[tid] = s1[tid];

  for (int jt = 0; jt < NB; jt += 64) {
    const int cur = (jt >> 6) & 1;
    wait_async0();           // my async writes into buf[cur] are in LDS
    __syncthreads();         // everyone's are visible; buf[cur^1] free for reuse

    if (jt + 64 < NB) {      // overlap: DMA next tile while computing this one
      const unsigned ndst = cur ? hb0o : hb1o;
      float* const s1n = cur ? s1c0 : s1c1;
      const unsigned goff = (unsigned)((size_t)(jt + 64) * HH * 2);
      for (int e = tid; e < (int)(HB_BYTES / 16); e += 256)
        async_ld_b128(ndst + (unsigned)e * 16u, goff + (unsigned)e * 16u,
                      (const void*)hidb);
      if (tid < 64) s1n[tid] = s1[jt + 64 + tid];
    }

    const unsigned short* Hb = cur ? Hb1 : Hb0;
    const float* s1c = cur ? s1c1 : s1c0;

#pragma unroll
    for (int sub = 0; sub < 2; ++sub) {
      const int j0 = sub * 32;
      Frag af;   // P tile 16x32: probabilities built in-register
#pragma unroll
      for (int v = 0; v < 8; ++v) {
        int kb = a_kbase(v, half);
        float p1 = __expf(leaky(s2i + s1c[j0 + kb]) - mi) * li;
        float p2 = __expf(leaky(s2i + s1c[j0 + kb + 1]) - mi) * li;
        af.u[v] = (unsigned int)f2bf_u16(p1) | ((unsigned int)f2bf_u16(p2) << 16);
      }
#pragma unroll
      for (int nt = 0; nt < 8; ++nt) {
        Frag bf;
#pragma unroll
        for (int v = 0; v < 8; ++v) {
          int K = half * 16 + 2 * v;
          unsigned int lo = Hb[(j0 + K) * HH + nt * 16 + l16];
          unsigned int hi = Hb[(j0 + K + 1) * HH + nt * 16 + l16];
          bf.u[v] = lo | (hi << 16);
        }
        acc[nt] = __builtin_amdgcn_wmma_f32_16x16x32_bf16(
            false, af.v, false, bf.v, (short)0, acc[nt], false, false);
      }
    }
  }

  // h2 = att@hidden + hidden  (residual) -> LDS for the FC A transpose
#pragma unroll
  for (int nt = 0; nt < 8; ++nt)
#pragma unroll
    for (int r = 0; r < 8; ++r) {
      int rr = r + 8 * half, cc = nt * 16 + l16;
      h2s[rr * HH + cc] = acc[nt][r] + hid[(size_t)(i0 + rr) * HH + cc];
    }
  __syncthreads();

  // h3 = leaky(h2 @ Wfc^T + bfc) via WMMA
  v8f accf[8];
#pragma unroll
  for (int nt = 0; nt < 8; ++nt) {
    float b = bfc[nt * 16 + l16];
#pragma unroll
    for (int r = 0; r < 8; ++r) accf[nt][r] = b;
  }
#pragma unroll
  for (int kf = 0; kf < 4; ++kf) {
    Frag af;
#pragma unroll
    for (int v = 0; v < 8; ++v) {
      int k = kf * 32 + a_kbase(v, half);
      af.u[v] = (unsigned int)f2bf_u16(h2s[l16 * HH + k]) |
                ((unsigned int)f2bf_u16(h2s[l16 * HH + k + 1]) << 16);
    }
#pragma unroll
    for (int nt = 0; nt < 8; ++nt) {
      Frag bf;
      const unsigned short* bp = &Wfc_s[(size_t)(nt * 16 + l16) * HH + kf * 32 + half * 16];
#pragma unroll
      for (int v = 0; v < 8; ++v) bf.u[v] = *(const unsigned int*)&bp[2 * v];
      accf[nt] = __builtin_amdgcn_wmma_f32_16x16x32_bf16(
          false, af.v, false, bf.v, (short)0, accf[nt], false, false);
    }
  }

  // out = sigmoid(h3 . Wout + bout) with wave32 shuffle reduction
  float wl[8];
#pragma unroll
  for (int nt = 0; nt < 8; ++nt) wl[nt] = Wout[nt * 16 + l16];
  const float bo = bout[0];
#pragma unroll
  for (int r = 0; r < 8; ++r) {
    float p = 0.f;
#pragma unroll
    for (int nt = 0; nt < 8; ++nt) p += leaky(accf[nt][r]) * wl[nt];
    for (int m = 8; m >= 1; m >>= 1) p += __shfl_xor(p, m, 16);
    if (l16 == 0) out[i0 + r + 8 * half] = sigm(p + bo);
  }
}

// ---------------- launch ----------------
extern "C" void kernel_launch(void* const* d_in, const int* in_sizes, int n_in,
                              void* d_out, int out_size, void* d_ws, size_t ws_size,
                              hipStream_t stream) {
  const float* x    = (const float*)d_in[0];
  const float* g1   = (const float*)d_in[1];
  const float* b1   = (const float*)d_in[2];
  const float* Wih  = (const float*)d_in[3];
  const float* Whh  = (const float*)d_in[4];
  const float* bih  = (const float*)d_in[5];
  const float* bhh  = (const float*)d_in[6];
  const float* g2   = (const float*)d_in[7];
  const float* b2   = (const float*)d_in[8];
  const float* Wt   = (const float*)d_in[9];
  const float* bt   = (const float*)d_in[10];
  const float* av   = (const float*)d_in[11];
  const float* Wfc  = (const float*)d_in[12];
  const float* bfc  = (const float*)d_in[13];
  const float* Wout = (const float*)d_in[14];
  const float* bout = (const float*)d_in[15];
  char* ws = (char*)d_ws;
  float* out = (float*)d_out;

  (void)hipFuncSetAttribute(reinterpret_cast<const void*>(lstm_kernel),
                            hipFuncAttributeMaxDynamicSharedMemorySize, (int)LSTM_LDS);
  (void)hipFuncSetAttribute(reinterpret_cast<const void*>(attn_out_kernel),
                            hipFuncAttributeMaxDynamicSharedMemorySize, (int)ATTN_LDS);

  bn1_stats_kernel<<<DD, 256, 0, stream>>>(x, g1, b1, ws);
  prep_kernel<<<128, 256, 0, stream>>>(Wih, Whh, bih, bhh, Wt, bt, av, Wfc, ws);
  lstm_kernel<<<NB / 16, 256, LSTM_LDS, stream>>>(x, ws);
  bn2_stats_kernel<<<HH, 256, 0, stream>>>(g2, b2, ws);
  rownorm_kernel<<<NB, 128, 0, stream>>>(ws);
  softmax_stats_kernel<<<NB / 256, 256, 0, stream>>>(ws);
  attn_out_kernel<<<NB / (8 * 16), 256, ATTN_LDS, stream>>>(bfc, Wout, bout, ws, out);
}
